// CrossAttention_7370163880742
// MI455X (gfx1250) — matrix operand order repaired and checked by reading, not compiled, not run on hardware
//
#include <hip/hip_runtime.h>
#include <hip/hip_bf16.h>
#include <stdint.h>

// ---------------------------------------------------------------------------
// CDNA5 (gfx1250) implementation of the cross-attention block.
// GEMMs + attention run on v_wmma_f32_16x16x32_f16 (wave32); global->LDS tile
// staging uses GLOBAL_LOAD_ASYNC_TO_LDS_B128 with double buffering (ASYNCcnt).
// ---------------------------------------------------------------------------

typedef __attribute__((ext_vector_type(16))) _Float16 v16h;
typedef __attribute__((ext_vector_type(8)))  float    v8f;
typedef _Float16 half_t;

#define DIM   512
#define HEADS 8
#define HD    64
#define DFF   2048
#define BATCH 2
#define SEQ   1024
#define M1    (BATCH * SEQ)        // 2048 tokens per stream
#define M2    (BATCH * 2 * SEQ)    // 4096 tokens in the fused stream
#define LN_EPS 1e-5f

// ---------------------------------------------------------------------------
// WMMA helper: D = A(16x32 f16) * B(32x16 f16) + C(16x16 f32)
// ---------------------------------------------------------------------------
__device__ __forceinline__ v8f wmma_16x16x32(v16h a, v16h b, v8f c) {
  return __builtin_amdgcn_wmma_f32_16x16x32_f16(
      /*neg_a=*/false, a, /*neg_b=*/false, b,
      /*c_mod=*/(short)0, c, /*reuse_a=*/false, /*reuse_b=*/false);
}

// Async 16-byte global -> LDS copy (per lane). Tracked by ASYNCcnt.
__device__ __forceinline__ void async_g2l_b128(void* lds, const void* gptr) {
  asm volatile("global_load_async_to_lds_b128 %0, %1, off"
               :: "v"((unsigned)(uintptr_t)lds), "v"(gptr)
               : "memory");
}
#define WAIT_ASYNC(n) asm volatile("s_wait_asynccnt " #n ::: "memory")

// A-fragment (16x32, M x K), source row-major with leading dim ld (halves).
// Order used here: lane L holds row M=L%16; K = 16*(L/16) + j (contiguous, matches B).
__device__ __forceinline__ v16h load_a_frag(const half_t* base, int ld, int lane) {
  const int m    = lane & 15;
  const int koff = (lane >> 4) * 16;
  const half_t* p = base + (size_t)m * ld + koff;
  v16h a;
#pragma unroll
  for (int j = 0; j < 8; ++j) { a[j] = p[j]; a[j + 8] = p[j + 8]; }
  return a;
}

// B-fragment (32x16, K x N) from an N-major source: src[n][k], ld in halves.
// Order used here: lane L holds column N=L%16, K = 16*(L/16) + j (contiguous).
__device__ __forceinline__ v16h load_bT_frag(const half_t* base, int ld, int lane) {
  const int n    = lane & 15;
  const int koff = (lane >> 4) * 16;
  const half_t* p = base + (size_t)n * ld + koff;
  v16h b;
#pragma unroll
  for (int j = 0; j < 16; ++j) b[j] = p[j];
  return b;
}

// B-fragment from a K-major source: src[k][n], ld in halves (P @ V product).
__device__ __forceinline__ v16h load_b_col_frag(const half_t* base, int ld,
                                                int ncol, int lane) {
  const int n    = (lane & 15) + ncol;
  const int koff = (lane >> 4) * 16;
  v16h b;
#pragma unroll
  for (int j = 0; j < 16; ++j) b[j] = base[(size_t)(koff + j) * ld + n];
  return b;
}

// ---------------------------------------------------------------------------
// GEMM: C[M,N] = A[M,K] (f16) * W[N,K]^T (f16) + bias (f32), optional ReLU.
// Block = 128 threads (4 waves); C tile 128x64, K-step 32.
// Wave w computes rows [w*32, w*32+32) x 64 cols (8 C fragments, 8 WMMA/iter).
// Double-buffered async global->LDS staging: 6 async ops per wave per tile.
// ---------------------------------------------------------------------------
#define GBM 128
#define GBN 64
#define GBK 32

__global__ void __launch_bounds__(128)
gemm_bias_kernel(const half_t* __restrict__ A, const half_t* __restrict__ W,
                 const float* __restrict__ bias,
                 float* __restrict__ C32, half_t* __restrict__ C16,
                 int M, int N, int K, int relu) {
  __shared__ half_t As[2][GBM][GBK];
  __shared__ half_t Bs[2][GBN][GBK];

  const int tid  = threadIdx.x;
  const int lane = tid & 31;
  const int wave = tid >> 5;
  const int m0   = blockIdx.x * GBM;
  const int n0   = blockIdx.y * GBN;

  v8f acc[2][4] = {};

  // Stage one K-tile into LDS buffer `buf` (6 async instrs per wave).
  auto stage = [&](int k0, int buf) {
#pragma unroll
    for (int i = tid; i < (GBM * GBK) / 8; i += 128) {       // 512 chunks
      const int r = i >> 2, c8 = (i & 3) << 3;
      async_g2l_b128(&As[buf][r][c8], A + (size_t)(m0 + r) * K + k0 + c8);
    }
#pragma unroll
    for (int i = tid; i < (GBN * GBK) / 8; i += 128) {       // 256 chunks
      const int r = i >> 2, c8 = (i & 3) << 3;
      async_g2l_b128(&Bs[buf][r][c8], W + (size_t)(n0 + r) * K + k0 + c8);
    }
  };

  stage(0, 0);
  int buf = 0;
  for (int k0 = 0; k0 < K; k0 += GBK) {
    if (k0 + GBK < K) {
      stage(k0 + GBK, buf ^ 1);   // prefetch next tile into the other buffer
      WAIT_ASYNC(0x6);            // current tile (older 6 ops) complete
    } else {
      WAIT_ASYNC(0x0);
    }
    __syncthreads();

    v16h bfr[4];
#pragma unroll
    for (int nf = 0; nf < 4; ++nf)
      bfr[nf] = load_bT_frag(&Bs[buf][nf * 16][0], GBK, lane);
    const v16h a0 = load_a_frag(&As[buf][wave * 32][0],      GBK, lane);
    const v16h a1 = load_a_frag(&As[buf][wave * 32 + 16][0], GBK, lane);
#pragma unroll
    for (int nf = 0; nf < 4; ++nf) acc[0][nf] = wmma_16x16x32(a0, bfr[nf], acc[0][nf]);
#pragma unroll
    for (int nf = 0; nf < 4; ++nf) acc[1][nf] = wmma_16x16x32(a1, bfr[nf], acc[1][nf]);

    __syncthreads();              // all reads done before next-tile staging
    buf ^= 1;
  }

  // Epilogue: C layout -> lane holds col n = lane%16, rows r + 8*(lane/16).
  const int ncl = lane & 15;
#pragma unroll
  for (int half = 0; half < 2; ++half) {
    const int gm_base = m0 + wave * 32 + half * 16 + ((lane >> 4) << 3);
#pragma unroll
    for (int nf = 0; nf < 4; ++nf) {
      const int gn = n0 + nf * 16 + ncl;
      const float bv = bias ? bias[gn] : 0.f;
#pragma unroll
      for (int r = 0; r < 8; ++r) {
        float v = acc[half][nf][r] + bv;
        if (relu) v = fmaxf(v, 0.f);
        const size_t idx = (size_t)(gm_base + r) * N + gn;
        if (C32) C32[idx] = v;
        if (C16) C16[idx] = (half_t)v;
      }
    }
  }
}

// ---------------------------------------------------------------------------
// Flash attention. q/k/v: [Btok, DIM] f16, head h in cols [h*64, h*64+64).
// Grid: (S_/64, HEADS, BATCH); block = 128 (4 waves; wave = 16 query rows).
// K/V tiles (32 keys) double-buffered via async LDS loads (4 ops/wave/tile).
// ---------------------------------------------------------------------------
__global__ void __launch_bounds__(128)
flash_attn_kernel(const half_t* __restrict__ qh, const half_t* __restrict__ kh,
                  const half_t* __restrict__ vh, const int* __restrict__ mask,
                  half_t* __restrict__ out, int S_) {
  __shared__ half_t Ks[2][32][HD];
  __shared__ half_t Vs[2][32][HD];
  __shared__ half_t Ps[4][16][32];
  __shared__ float  mflag[2][32];

  const int tid  = threadIdx.x;
  const int lane = tid & 31;
  const int wave = tid >> 5;
  const int h    = blockIdx.y;
  const int b    = blockIdx.z;
  const int q0   = b * S_ + blockIdx.x * 64 + wave * 16;   // global query row
  const int kv0  = b * S_;
  const int* mrow = mask + (size_t)b * S_;
  const float scale = 0.125f;                               // 1/sqrt(64)

  // Q fragments held in registers for the whole pass (K = 64 -> 2 frags).
  const half_t* qbase = qh + (size_t)q0 * DIM + h * HD;
  const v16h qa0 = load_a_frag(qbase,      DIM, lane);
  const v16h qa1 = load_a_frag(qbase + 32, DIM, lane);

  v8f   O[4] = {};
  float m_run[8], l_run[8];
#pragma unroll
  for (int r = 0; r < 8; ++r) { m_run[r] = -3.0e38f; l_run[r] = 0.f; }

  auto stage = [&](int kt, int buf) {
#pragma unroll
    for (int i = tid; i < 256; i += 128) {                  // 32x64 halves
      const int r = i >> 3, c8 = (i & 7) << 3;
      async_g2l_b128(&Ks[buf][r][c8],
                     kh + (size_t)(kv0 + kt + r) * DIM + h * HD + c8);
    }
#pragma unroll
    for (int i = tid; i < 256; i += 128) {
      const int r = i >> 3, c8 = (i & 7) << 3;
      async_g2l_b128(&Vs[buf][r][c8],
                     vh + (size_t)(kv0 + kt + r) * DIM + h * HD + c8);
    }
    if (tid < 32) mflag[buf][tid] = (mrow[kt + tid] == 1) ? 1.f : 0.f;
  };

  stage(0, 0);
  int buf = 0;
  for (int kt = 0; kt < S_; kt += 32) {
    if (kt + 32 < S_) {
      stage(kt + 32, buf ^ 1);
      WAIT_ASYNC(0x4);
    } else {
      WAIT_ASYNC(0x0);
    }
    __syncthreads();

    // Scores: S = Q(16x64) @ K_tile^T(64x32), as two 16x16 C frags.
    v8f s0 = {}, s1 = {};
    s0 = wmma_16x16x32(qa0, load_bT_frag(&Ks[buf][0][0],       HD, lane), s0);
    s0 = wmma_16x16x32(qa1, load_bT_frag(&Ks[buf][0][0] + 32,  HD, lane), s0);
    s1 = wmma_16x16x32(qa0, load_bT_frag(&Ks[buf][16][0],      HD, lane), s1);
    s1 = wmma_16x16x32(qa1, load_bT_frag(&Ks[buf][16][0] + 32, HD, lane), s1);

    const float f0 = mflag[buf][lane & 15];
    const float f1 = mflag[buf][16 + (lane & 15)];
    float corr[8];
#pragma unroll
    for (int r = 0; r < 8; ++r) {
      float a0 = (f0 != 0.f) ? s0[r] * scale : -1.0e9f;
      float a1 = (f1 != 0.f) ? s1[r] * scale : -1.0e9f;
      // Row max across the 16 columns (xor masks stay inside lane halves).
      float mx = fmaxf(a0, a1);
#pragma unroll
      for (int off = 8; off > 0; off >>= 1) mx = fmaxf(mx, __shfl_xor(mx, off));
      const float mnew = fmaxf(m_run[r], mx);
      corr[r] = __expf(m_run[r] - mnew);
      m_run[r] = mnew;
      const float p0 = __expf(a0 - mnew);
      const float p1 = __expf(a1 - mnew);
      float ps = p0 + p1;
#pragma unroll
      for (int off = 8; off > 0; off >>= 1) ps += __shfl_xor(ps, off);
      l_run[r] = l_run[r] * corr[r] + ps;
      // Stash P (C layout -> LDS) for re-load in A layout.
      const int row = r + ((lane >> 4) << 3);
      Ps[wave][row][lane & 15]        = (half_t)p0;
      Ps[wave][row][16 + (lane & 15)] = (half_t)p1;
    }
#pragma unroll
    for (int no = 0; no < 4; ++no)
#pragma unroll
      for (int r = 0; r < 8; ++r) O[no][r] *= corr[r];
    __syncthreads();                       // Ps ready

    // O += P(16x32) @ V_tile(32x64)
    const v16h pa = load_a_frag(&Ps[wave][0][0], 32, lane);
#pragma unroll
    for (int no = 0; no < 4; ++no) {
      const v16h vb = load_b_col_frag(&Vs[buf][0][0], HD, no * 16, lane);
      O[no] = wmma_16x16x32(pa, vb, O[no]);
    }
    __syncthreads();                       // buffer reads done before re-stage
    buf ^= 1;
  }

  // Normalize and store (f16 -> feeds the Wo GEMM).
  float inv[8];
#pragma unroll
  for (int r = 0; r < 8; ++r) inv[r] = 1.0f / l_run[r];
  const int ncl   = lane & 15;
  const int gmrow = q0 + ((lane >> 4) << 3);
#pragma unroll
  for (int no = 0; no < 4; ++no)
#pragma unroll
    for (int r = 0; r < 8; ++r)
      out[(size_t)(gmrow + r) * DIM + h * HD + no * 16 + ncl] =
          (half_t)(O[no][r] * inv[r]);
}

// ---------------------------------------------------------------------------
// Fused residual + LayerNorm. One wave per token (DIM = 512 = 16 per lane).
// ---------------------------------------------------------------------------
__global__ void __launch_bounds__(128)
add_ln_kernel(const float* __restrict__ X, const float* __restrict__ R,
              const float* __restrict__ g, const float* __restrict__ bt,
              float* __restrict__ o32, half_t* __restrict__ o16, int Mtok) {
  const int lane = threadIdx.x & 31;
  const int wave = threadIdx.x >> 5;
  const int t = blockIdx.x * 4 + wave;
  if (t >= Mtok) return;
  const float* px = X + (size_t)t * DIM;
  const float* pr = R ? R + (size_t)t * DIM : nullptr;

  float v[16];
  float s = 0.f;
#pragma unroll
  for (int i = 0; i < 16; ++i) {
    const int c = i * 32 + lane;
    float x = px[c] + (pr ? pr[c] : 0.f);
    v[i] = x; s += x;
  }
#pragma unroll
  for (int off = 16; off > 0; off >>= 1) s += __shfl_xor(s, off);
  const float mean = s * (1.f / (float)DIM);
  float q = 0.f;
#pragma unroll
  for (int i = 0; i < 16; ++i) { const float d = v[i] - mean; q += d * d; }
#pragma unroll
  for (int off = 16; off > 0; off >>= 1) q += __shfl_xor(q, off);
  const float rstd = rsqrtf(q * (1.f / (float)DIM) + LN_EPS);
#pragma unroll
  for (int i = 0; i < 16; ++i) {
    const int c = i * 32 + lane;
    const float y = (v[i] - mean) * rstd * g[c] + bt[c];
    o32[(size_t)t * DIM + c] = y;
    if (o16) o16[(size_t)t * DIM + c] = (half_t)y;
  }
}

// ---------------------------------------------------------------------------
// Glue kernels.
// ---------------------------------------------------------------------------
__global__ void f32_to_f16_kernel(const float* __restrict__ in,
                                  half_t* __restrict__ out, int n) {
  const int i = blockIdx.x * blockDim.x + threadIdx.x;
  if (i < n) out[i] = (half_t)in[i];
}

// x0[b, p, :] = sem_out[b, p] (p < S) else emo_out[b, p-S]; f32 + f16 copies.
__global__ void build_x0_kernel(const float* __restrict__ semo,
                                const float* __restrict__ emoo,
                                float* __restrict__ x32,
                                half_t* __restrict__ x16) {
  const int i = blockIdx.x * blockDim.x + threadIdx.x;
  if (i >= M2 * DIM) return;
  const int t = i >> 9, c = i & 511;
  const int b = t >> 11, p = t & 2047;
  const float v = (p < SEQ)
      ? semo[((size_t)(b * SEQ + p) << 9) + c]
      : emoo[((size_t)(b * SEQ + (p - SEQ)) << 9) + c];
  x32[i] = v;
  x16[i] = (half_t)v;
}

__global__ void build_tmask_kernel(const int* __restrict__ em,
                                   const int* __restrict__ sm,
                                   int* __restrict__ tm) {
  const int i = blockIdx.x * blockDim.x + threadIdx.x;
  if (i >= M2) return;
  const int b = i >> 11, p = i & 2047;
  tm[i] = (p < SEQ) ? em[b * SEQ + p] : sm[b * SEQ + (p - SEQ)];
}

// out[b, s, c] = x2[b*2S + (c<512 ? s : S+s)][c&511] * emo_mask[b, s]
__global__ void finalize_kernel(const float* __restrict__ x2,
                                const int* __restrict__ emask,
                                float* __restrict__ out) {
  const int i = blockIdx.x * blockDim.x + threadIdx.x;
  if (i >= BATCH * SEQ * 2 * DIM) return;
  const int b = i >> 20, rem = i & 1048575;
  const int s = rem >> 10, c = rem & 1023;
  const int p = (c < DIM) ? s : (SEQ + s);
  const float v = x2[((size_t)(b * 2 * SEQ + p) << 9) + (c & 511)];
  out[i] = (emask[b * SEQ + s] == 1) ? v : 0.f;
}

// ---------------------------------------------------------------------------
// Host-side launch helpers.
// ---------------------------------------------------------------------------
static inline void conv16(hipStream_t s, const float* in, half_t* out, int n) {
  f32_to_f16_kernel<<<dim3((n + 255) / 256), 256, 0, s>>>(in, out, n);
}
static inline void gemm(hipStream_t s, const half_t* A, const half_t* W,
                        const float* bias, float* c32, half_t* c16,
                        int M, int N, int K, int relu) {
  gemm_bias_kernel<<<dim3(M / GBM, N / GBN), 128, 0, s>>>(A, W, bias, c32, c16,
                                                          M, N, K, relu);
}
static inline void attn(hipStream_t s, const half_t* q, const half_t* k,
                        const half_t* v, const int* mask, half_t* o, int S_) {
  flash_attn_kernel<<<dim3(S_ / 64, HEADS, BATCH), 128, 0, s>>>(q, k, v, mask,
                                                                o, S_);
}

extern "C" void kernel_launch(void* const* d_in, const int* in_sizes, int n_in,
                              void* d_out, int out_size, void* d_ws,
                              size_t ws_size, hipStream_t stream) {
  (void)in_sizes; (void)n_in; (void)out_size; (void)ws_size;

  // ---- inputs (setup_inputs dict order) ----
  const float* emo_x   = (const float*)d_in[0];
  const float* sem_x   = (const float*)d_in[1];
  const int*   emo_m   = (const int*)d_in[2];
  const int*   sem_m   = (const int*)d_in[3];
  const float* Wq_emo  = (const float*)d_in[4];   const float* bq_emo = (const float*)d_in[5];
  const float* Wk_emo  = (const float*)d_in[6];   const float* bk_emo = (const float*)d_in[7];
  const float* Wv_emo  = (const float*)d_in[8];   const float* bv_emo = (const float*)d_in[9];
  const float* Wq_sem  = (const float*)d_in[10];  const float* bq_sem = (const float*)d_in[11];
  const float* Wk_sem  = (const float*)d_in[12];  const float* bk_sem = (const float*)d_in[13];
  const float* Wv_sem  = (const float*)d_in[14];  const float* bv_sem = (const float*)d_in[15];
  const float* sem_Wi  = (const float*)d_in[16];  const float* sem_bi = (const float*)d_in[17];
  const float* sem_Wo  = (const float*)d_in[18];  const float* sem_bo = (const float*)d_in[19];
  const float* emo_Wi  = (const float*)d_in[20];  const float* emo_bi = (const float*)d_in[21];
  const float* emo_Wo  = (const float*)d_in[22];  const float* emo_bo = (const float*)d_in[23];
  const float* t_Wi    = (const float*)d_in[24];  const float* t_bi   = (const float*)d_in[25];
  const float* t_Wo    = (const float*)d_in[26];  const float* t_bo   = (const float*)d_in[27];
  const float* t_W1    = (const float*)d_in[28];  const float* t_b1   = (const float*)d_in[29];
  const float* t_W2    = (const float*)d_in[30];  const float* t_b2   = (const float*)d_in[31];
  const float* ln1_g   = (const float*)d_in[32];  const float* ln1_b  = (const float*)d_in[33];
  const float* ln2_g   = (const float*)d_in[34];  const float* ln2_b  = (const float*)d_in[35];

  // ---- workspace bump allocator (256B aligned) ----
  char* base = (char*)d_ws;
  size_t cur = 0;
  auto alloc = [&](size_t bytes) -> void* {
    cur = (cur + 255) & ~(size_t)255;
    void* p = base + cur;
    cur += bytes;
    return p;
  };
  const size_t DD = (size_t)DIM * DIM;   // 512*512

  // f16 weights
  half_t* hWq_emo = (half_t*)alloc(DD * 2);
  half_t* hWk_emo = (half_t*)alloc(DD * 2);
  half_t* hWv_emo = (half_t*)alloc(DD * 2);
  half_t* hWq_sem = (half_t*)alloc(DD * 2);
  half_t* hWk_sem = (half_t*)alloc(DD * 2);
  half_t* hWv_sem = (half_t*)alloc(DD * 2);
  half_t* hWi_sem = (half_t*)alloc(3 * DD * 2);
  half_t* hWo_sem = (half_t*)alloc(DD * 2);
  half_t* hWi_emo = (half_t*)alloc(3 * DD * 2);
  half_t* hWo_emo = (half_t*)alloc(DD * 2);
  half_t* hWi_t   = (half_t*)alloc(3 * DD * 2);
  half_t* hWo_t   = (half_t*)alloc(DD * 2);
  half_t* hW1     = (half_t*)alloc((size_t)DFF * DIM * 2);
  half_t* hW2     = (half_t*)alloc((size_t)DIM * DFF * 2);

  // f16 activations
  const size_t A1 = (size_t)M1 * DIM, A2 = (size_t)M2 * DIM;
  half_t* hx_emo  = (half_t*)alloc(A1 * 2);
  half_t* hx_sem  = (half_t*)alloc(A1 * 2);
  half_t* q_emo16 = (half_t*)alloc(A1 * 2);
  half_t* k_emo16 = (half_t*)alloc(A1 * 2);
  half_t* v_emo16 = (half_t*)alloc(A1 * 2);
  half_t* q_sem16 = (half_t*)alloc(A1 * 2);
  half_t* k_sem16 = (half_t*)alloc(A1 * 2);
  half_t* v_sem16 = (half_t*)alloc(A1 * 2);
  half_t* qh_sem  = (half_t*)alloc(A1 * 2);
  half_t* kh_sem  = (half_t*)alloc(A1 * 2);
  half_t* vh_sem  = (half_t*)alloc(A1 * 2);
  half_t* qh_emo  = (half_t*)alloc(A1 * 2);
  half_t* kh_emo  = (half_t*)alloc(A1 * 2);
  half_t* vh_emo  = (half_t*)alloc(A1 * 2);
  half_t* o_sem   = (half_t*)alloc(A1 * 2);
  half_t* o_emo   = (half_t*)alloc(A1 * 2);
  float*  semo32  = (float*)alloc(A1 * 4);
  float*  emoo32  = (float*)alloc(A1 * 4);
  float*  x0_32   = (float*)alloc(A2 * 4);
  half_t* x0_16   = (half_t*)alloc(A2 * 2);
  int*    tmask   = (int*)alloc((size_t)M2 * 4);
  half_t* q_t     = (half_t*)alloc(A2 * 2);
  half_t* k_t     = (half_t*)alloc(A2 * 2);
  half_t* v_t     = (half_t*)alloc(A2 * 2);
  half_t* o_t     = (half_t*)alloc(A2 * 2);
  float*  y_t32   = (float*)alloc(A2 * 4);
  float*  x1_32   = (float*)alloc(A2 * 4);
  half_t* x1_16   = (half_t*)alloc(A2 * 2);
  half_t* ffh     = (half_t*)alloc((size_t)M2 * DFF * 2);
  float*  y2_32   = (float*)alloc(A2 * 4);
  float*  x2_32   = (float*)alloc(A2 * 4);

  // ---- 0) f32 -> f16 conversions ----
  conv16(stream, Wq_emo, hWq_emo, DD);  conv16(stream, Wk_emo, hWk_emo, DD);
  conv16(stream, Wv_emo, hWv_emo, DD);  conv16(stream, Wq_sem, hWq_sem, DD);
  conv16(stream, Wk_sem, hWk_sem, DD);  conv16(stream, Wv_sem, hWv_sem, DD);
  conv16(stream, sem_Wi, hWi_sem, 3 * DD); conv16(stream, sem_Wo, hWo_sem, DD);
  conv16(stream, emo_Wi, hWi_emo, 3 * DD); conv16(stream, emo_Wo, hWo_emo, DD);
  conv16(stream, t_Wi,   hWi_t,   3 * DD); conv16(stream, t_Wo,   hWo_t,   DD);
  conv16(stream, t_W1, hW1, DFF * DIM);    conv16(stream, t_W2, hW2, DIM * DFF);
  conv16(stream, emo_x, hx_emo, A1);       conv16(stream, sem_x, hx_sem, A1);

  // ---- 1) outer Q/K/V projections ----
  gemm(stream, hx_emo, hWq_emo, bq_emo, nullptr, q_emo16, M1, DIM, DIM, 0);
  gemm(stream, hx_emo, hWk_emo, bk_emo, nullptr, k_emo16, M1, DIM, DIM, 0);
  gemm(stream, hx_emo, hWv_emo, bv_emo, nullptr, v_emo16, M1, DIM, DIM, 0);
  gemm(stream, hx_sem, hWq_sem, bq_sem, nullptr, q_sem16, M1, DIM, DIM, 0);
  gemm(stream, hx_sem, hWk_sem, bk_sem, nullptr, k_sem16, M1, DIM, DIM, 0);
  gemm(stream, hx_sem, hWv_sem, bv_sem, nullptr, v_sem16, M1, DIM, DIM, 0);

  // ---- 2) inner Wi projections (row-slices of Wi are Wq/Wk/Wv) ----
  gemm(stream, q_emo16, hWi_sem,          sem_bi,           nullptr, qh_sem, M1, DIM, DIM, 0);
  gemm(stream, k_sem16, hWi_sem + DD,     sem_bi + DIM,     nullptr, kh_sem, M1, DIM, DIM, 0);
  gemm(stream, v_sem16, hWi_sem + 2 * DD, sem_bi + 2 * DIM, nullptr, vh_sem, M1, DIM, DIM, 0);
  gemm(stream, q_sem16, hWi_emo,          emo_bi,           nullptr, qh_emo, M1, DIM, DIM, 0);
  gemm(stream, k_emo16, hWi_emo + DD,     emo_bi + DIM,     nullptr, kh_emo, M1, DIM, DIM, 0);
  gemm(stream, v_emo16, hWi_emo + 2 * DD, emo_bi + 2 * DIM, nullptr, vh_emo, M1, DIM, DIM, 0);

  // ---- 3) cross attentions ----
  attn(stream, qh_sem, kh_sem, vh_sem, sem_m, o_sem, SEQ);
  attn(stream, qh_emo, kh_emo, vh_emo, emo_m, o_emo, SEQ);

  // ---- 4) Wo projections + assemble x0 = concat(sem_out, emo_out) ----
  gemm(stream, o_sem, hWo_sem, sem_bo, semo32, nullptr, M1, DIM, DIM, 0);
  gemm(stream, o_emo, hWo_emo, emo_bo, emoo32, nullptr, M1, DIM, DIM, 0);
  build_x0_kernel<<<dim3((M2 * DIM + 255) / 256), 256, 0, stream>>>(
      semo32, emoo32, x0_32, x0_16);
  build_tmask_kernel<<<dim3((M2 + 255) / 256), 256, 0, stream>>>(emo_m, sem_m,
                                                                 tmask);

  // ---- 5) fused self-attention (t block) ----
  gemm(stream, x0_16, hWi_t,          t_bi,           nullptr, q_t, M2, DIM, DIM, 0);
  gemm(stream, x0_16, hWi_t + DD,     t_bi + DIM,     nullptr, k_t, M2, DIM, DIM, 0);
  gemm(stream, x0_16, hWi_t + 2 * DD, t_bi + 2 * DIM, nullptr, v_t, M2, DIM, DIM, 0);
  attn(stream, q_t, k_t, v_t, tmask, o_t, 2 * SEQ);
  gemm(stream, o_t, hWo_t, t_bo, y_t32, nullptr, M2, DIM, DIM, 0);

  // ---- 6) residual + LN1 ----
  add_ln_kernel<<<dim3(M2 / 4), 128, 0, stream>>>(x0_32, y_t32, ln1_g, ln1_b,
                                                  x1_32, x1_16, M2);

  // ---- 7) FFN ----
  gemm(stream, x1_16, hW1, t_b1, nullptr, ffh, M2, DFF, DIM, /*relu=*/1);
  gemm(stream, ffh,   hW2, t_b2, y2_32, nullptr, M2, DIM, DFF, 0);

  // ---- 8) residual + LN2 ----
  add_ln_kernel<<<dim3(M2 / 4), 128, 0, stream>>>(x1_32, y2_32, ln2_g, ln2_b,
                                                  x2_32, nullptr, M2);

  // ---- 9) feature concat + key mask ----
  finalize_kernel<<<dim3((BATCH * SEQ * 2 * DIM + 255) / 256), 256, 0, stream>>>(
      x2_32, emo_m, (float*)d_out);
}
